// TransformerLayer_39298950758497
// MI455X (gfx1250) — matrix operand order repaired and checked by reading, not compiled, not run on hardware
//
#include <hip/hip_runtime.h>
#include <hip/hip_bf16.h>
#include <stdint.h>

#define B_   2
#define H_   16
#define S_   4096
#define DM   1024
#define DFF  4096
#define DH   64
#define WIN  256
#define NC   (S_/WIN)
#define KSTEP 64

typedef __attribute__((ext_vector_type(16))) __bf16       v16bf;
typedef __attribute__((ext_vector_type(8)))  float        v8f;
typedef __attribute__((ext_vector_type(4)))  unsigned int v4u;
typedef __attribute__((ext_vector_type(4)))  int          v4i;

union FragBF { v16bf v; v4u q[2]; unsigned short u[16]; };

__device__ __forceinline__ unsigned short f2bf(float f) {
    unsigned int u = __float_as_uint(f);
    unsigned int r = u + 0x7FFFu + ((u >> 16) & 1u);   // RNE
    return (unsigned short)(r >> 16);
}

// ---------------------------------------------------------------------------
// CDNA5 async global->LDS copy (16B), ASYNCcnt-tracked. Guarded fallback.
// ---------------------------------------------------------------------------
#if defined(__AMDGCN__) && __has_builtin(__builtin_amdgcn_global_load_async_to_lds_b128)
#define HAVE_ASYNC_LDS 1
#else
#define HAVE_ASYNC_LDS 0
#endif

__device__ __forceinline__ void async_copy16(unsigned short* ldst,
                                             const unsigned short* gsrc) {
#if HAVE_ASYNC_LDS
    __builtin_amdgcn_global_load_async_to_lds_b128(
        (__attribute__((address_space(1))) v4i*)gsrc,
        (__attribute__((address_space(3))) v4i*)ldst,
        0, 0);
#else
    *(v4u*)ldst = *(const v4u*)gsrc;
#endif
}

__device__ __forceinline__ void wait_async0() {
#if HAVE_ASYNC_LDS
#if __has_builtin(__builtin_amdgcn_s_wait_asynccnt)
    __builtin_amdgcn_s_wait_asynccnt(0);
#else
    asm volatile("s_wait_asynccnt 0" ::: "memory");
#endif
#endif
}

// ---------------------------------------------------------------------------
// src (B,S,DM) fp32 -> qb[b,h,s,d] bf16 (rows contiguous in d for Q/K frags)
//                    -> vt[b,h,d,s] bf16 (contiguous in s for PV B-frags)
// ---------------------------------------------------------------------------
__global__ void prep_qkv(const float* __restrict__ src,
                         unsigned short* __restrict__ qb,
                         unsigned short* __restrict__ vt) {
    int i = blockIdx.x * 256 + threadIdx.x;        // over B*S*DM = 8388608
    int dmod = i % DM;
    int bs   = i / DM;
    int s    = bs % S_;
    int b    = bs / S_;
    int h = dmod / DH, d = dmod % DH;
    unsigned short v = f2bf(src[i]);
    long bh = (long)b * H_ + h;
    qb[(bh * S_ + s) * DH + d]        = v;
    vt[(bh * DH + d) * (long)S_ + s]  = v;
}

// ---------------------------------------------------------------------------
// Win (K x N, fp32, row-major) -> Wt (N x K, bf16)  [coalesced writes]
// ---------------------------------------------------------------------------
__global__ void transpose_bf16(const float* __restrict__ Win,
                               unsigned short* __restrict__ Wt,
                               int K, int N) {
    int i = blockIdx.x * 256 + threadIdx.x;        // over N*K
    int k = i % K;
    int n = i / K;
    Wt[i] = f2bf(Win[(long)k * N + n]);
}

// ---------------------------------------------------------------------------
// Sliding-window attention (flash-style online softmax, bf16 WMMA).
// Block: 512 threads = 16 waves; one block per (b, h, chunk of 256 queries).
// Wave w handles query rows [16w, 16w+16) of the chunk.
// ---------------------------------------------------------------------------
__global__ __launch_bounds__(512) void attn_kernel(
        const unsigned short* __restrict__ qb,
        const unsigned short* __restrict__ vt,
        float* __restrict__ attn) {
    __shared__ unsigned short pbuf[16 * 16 * 32];   // 16 waves * 16x32 bf16

    const int tid  = threadIdx.x;
    const int w    = tid >> 5;
    const int lane = tid & 31;
    const int hf   = lane >> 4;
    const int ln   = lane & 15;

    const int blk = blockIdx.x;           // b*H*NC + h*NC + c
    const int c   = blk % NC;
    const int bh  = blk / NC;
    const int b   = bh / H_;
    const int h   = bh % H_;

    const unsigned short* qbase = qb + (long)bh * S_ * DH;
    const unsigned short* vbase = vt + (long)bh * DH * S_;

    // Q A-fragments (row = lane ln, dims split K=0..31 / 32..63)
    const int qrow = c * WIN + 16 * w + ln;
    const unsigned short* qptr = qbase + (long)qrow * DH;
    FragBF qa0, qa1;
    qa0.q[0] = *(const v4u*)(qptr +  0 + 8 * hf);
    qa0.q[1] = *(const v4u*)(qptr + 16 + 8 * hf);
    qa1.q[0] = *(const v4u*)(qptr + 32 + 8 * hf);
    qa1.q[1] = *(const v4u*)(qptr + 48 + 8 * hf);

    v8f o[4] = {v8f{}, v8f{}, v8f{}, v8f{}};
    float m_[8], l_[8];
#pragma unroll
    for (int r = 0; r < 8; r++) { m_[r] = -1e9f; l_[r] = 0.f; }

    const int qmin   = c * WIN + 16 * w;
    const int kstart = c * WIN - WIN;
    unsigned short* pw = pbuf + w * 512;

    for (int g = 0; g < 24; ++g) {
        const int kb2 = kstart + 32 * g;
        if (kb2 + 32 <= 0 || kb2 >= S_) continue;                      // OOB group
        if (kb2 + 32 <= qmin - WIN || kb2 > qmin + 15 + WIN) continue; // out of window

        // ---- scores S = Q Kᵀ for keys [kb2, kb2+32) -------------------
        FragBF kf;
        v8f s0 = {}, s1 = {};
        const unsigned short* k0p = qbase + (long)(kb2 + ln) * DH;
        kf.q[0] = *(const v4u*)(k0p + 8 * hf);
        kf.q[1] = *(const v4u*)(k0p + 16 + 8 * hf);
        s0 = __builtin_amdgcn_wmma_f32_16x16x32_bf16(false, qa0.v, false, kf.v, (short)0, s0, false, false);
        kf.q[0] = *(const v4u*)(k0p + 32 + 8 * hf);
        kf.q[1] = *(const v4u*)(k0p + 48 + 8 * hf);
        s0 = __builtin_amdgcn_wmma_f32_16x16x32_bf16(false, qa1.v, false, kf.v, (short)0, s0, false, false);

        const unsigned short* k1p = qbase + (long)(kb2 + 16 + ln) * DH;
        kf.q[0] = *(const v4u*)(k1p + 8 * hf);
        kf.q[1] = *(const v4u*)(k1p + 16 + 8 * hf);
        s1 = __builtin_amdgcn_wmma_f32_16x16x32_bf16(false, qa0.v, false, kf.v, (short)0, s1, false, false);
        kf.q[0] = *(const v4u*)(k1p + 32 + 8 * hf);
        kf.q[1] = *(const v4u*)(k1p + 48 + 8 * hf);
        s1 = __builtin_amdgcn_wmma_f32_16x16x32_bf16(false, qa1.v, false, kf.v, (short)0, s1, false, false);

        // ---- masked online softmax update -----------------------------
        const int kp0 = kb2 + ln, kp1 = kb2 + 16 + ln;
#pragma unroll
        for (int r = 0; r < 8; r++) {
            const int qp = qmin + r + 8 * hf;
            const bool v0 = (kp0 - qp <= WIN) && (qp - kp0 <= WIN);
            const bool v1 = (kp1 - qp <= WIN) && (qp - kp1 <= WIN);
            float a0 = v0 ? s0[r] : -1e9f;
            float a1 = v1 ? s1[r] : -1e9f;
            float t = fmaxf(a0, a1);
            t = fmaxf(t, __shfl_xor(t, 1, 32));
            t = fmaxf(t, __shfl_xor(t, 2, 32));
            t = fmaxf(t, __shfl_xor(t, 4, 32));
            t = fmaxf(t, __shfl_xor(t, 8, 32));
            const float mn   = fmaxf(m_[r], t);
            const float corr = __expf(m_[r] - mn);
            const float p0 = v0 ? __expf(a0 - mn) : 0.f;
            const float p1 = v1 ? __expf(a1 - mn) : 0.f;
            float rs = p0 + p1;
            rs += __shfl_xor(rs, 1, 32);
            rs += __shfl_xor(rs, 2, 32);
            rs += __shfl_xor(rs, 4, 32);
            rs += __shfl_xor(rs, 8, 32);
            l_[r] = l_[r] * corr + rs;
            m_[r] = mn;
            o[0][r] *= corr; o[1][r] *= corr; o[2][r] *= corr; o[3][r] *= corr;
            const int row = r + 8 * hf;
            pw[row * 32 + ln]      = f2bf(p0);
            pw[row * 32 + 16 + ln] = f2bf(p1);
        }

        // C-layout -> A-layout re-fragmentation through LDS (wave-private)
        asm volatile("s_wait_dscnt 0" ::: "memory");
        FragBF pf;
        pf.q[0] = *(const v4u*)(pw + ln * 32 +  0 + 8 * hf);
        pf.q[1] = *(const v4u*)(pw + ln * 32 + 16 + 8 * hf);

        // ---- O += P V  (V^T prepacked: vt[d][s]) ----------------------
#pragma unroll
        for (int t = 0; t < 4; t++) {
            FragBF vf;
            const unsigned short* vp = vbase + (long)(16 * t + ln) * S_ + kb2 + 8 * hf;
            vf.q[0] = *(const v4u*)vp;
            vf.q[1] = *(const v4u*)(vp + 16);
            o[t] = __builtin_amdgcn_wmma_f32_16x16x32_bf16(false, pf.v, false, vf.v, (short)0, o[t], false, false);
        }
    }

    // ---- normalize + scatter to attn[b, s, h*64+d] ---------------------
#pragma unroll
    for (int r = 0; r < 8; r++) {
        const float inv = (l_[r] > 0.f) ? 1.f / l_[r] : 0.f;
        const int row = c * WIN + 16 * w + r + 8 * hf;
        float* op = attn + ((long)b * S_ + row) * DM + h * DH + ln;
        op[0]  = o[0][r] * inv;
        op[16] = o[1][r] * inv;
        op[32] = o[2][r] * inv;
        op[48] = o[3][r] * inv;
    }
}

// ---------------------------------------------------------------------------
// y = LayerNorm(A + Badd) * g + b ; optional bf16 copy. One block per row.
// ---------------------------------------------------------------------------
__global__ __launch_bounds__(256) void ln_kernel(
        const float* __restrict__ A, const float* __restrict__ Badd,
        const float* __restrict__ g, const float* __restrict__ bt,
        float* __restrict__ xout, unsigned short* __restrict__ xb16) {
    __shared__ float red[16];
    const int row = blockIdx.x;
    const float* pa = A    + (long)row * DM;
    const float* pb = Badd + (long)row * DM;

    float vals[4];
    float s = 0.f, s2 = 0.f;
#pragma unroll
    for (int i = 0; i < 4; i++) {
        const int col = threadIdx.x + 256 * i;
        const float v = pa[col] + pb[col];
        vals[i] = v; s += v; s2 += v * v;
    }
#pragma unroll
    for (int msk = 1; msk < 32; msk <<= 1) {
        s  += __shfl_xor(s,  msk, 32);
        s2 += __shfl_xor(s2, msk, 32);
    }
    const int w = threadIdx.x >> 5, lane = threadIdx.x & 31;
    if (lane == 0) { red[w] = s; red[8 + w] = s2; }
    __syncthreads();
    float ts = 0.f, ts2 = 0.f;
#pragma unroll
    for (int i = 0; i < 8; i++) { ts += red[i]; ts2 += red[8 + i]; }
    const float mu   = ts  * (1.f / DM);
    const float var  = ts2 * (1.f / DM) - mu * mu;
    const float rstd = rsqrtf(var + 1e-5f);
#pragma unroll
    for (int i = 0; i < 4; i++) {
        const int col = threadIdx.x + 256 * i;
        const float y = (vals[i] - mu) * rstd * g[col] + bt[col];
        xout[(long)row * DM + col] = y;
        if (xb16) xb16[(long)row * DM + col] = f2bf(y);
    }
}

// ---------------------------------------------------------------------------
// C = act(A @ Btᵀ + bias).  A: MxK bf16 row-major, Bt: NxK bf16 (pre-transposed).
// Block 256 = 8 waves; block tile 128x64; wave tile 16x64.
// K-panels (A:128x64, B:64x64) double-buffered in LDS via async global->LDS DMA
// (ASYNCcnt), overlapped with the 8 WMMAs of the resident panel.
// ---------------------------------------------------------------------------
__global__ __launch_bounds__(256) void gemm_bf16(
        const unsigned short* __restrict__ A,
        const unsigned short* __restrict__ Bt,
        const float* __restrict__ bias,
        float* __restrict__ Cf, unsigned short* __restrict__ Cb,
        int M, int N, int K, int relu) {
    __shared__ unsigned short smA[2][128 * KSTEP];   // 2 x 16 KB
    __shared__ unsigned short smB[2][64  * KSTEP];   // 2 x  8 KB

    const int tid  = threadIdx.x;
    const int w    = tid >> 5;
    const int lane = tid & 31;
    const int hf   = lane >> 4;
    const int ln   = lane & 15;

    const int rowt = blockIdx.y * 128;
    const int colt = blockIdx.x * 64;

    // Stage one K-panel: A = 1024 16B chunks (4/thread), B = 512 (2/thread).
    auto issue = [&](int kb, int buf) {
#pragma unroll
        for (int it = 0; it < 4; ++it) {
            const int j   = tid + 256 * it;
            const int row = j >> 3;
            const int kc  = (j & 7) * 8;
            async_copy16(&smA[buf][row * KSTEP + kc],
                         A + (long)(rowt + row) * K + kb + kc);
        }
#pragma unroll
        for (int it = 0; it < 2; ++it) {
            const int j   = tid + 256 * it;
            const int col = j >> 3;
            const int kc  = (j & 7) * 8;
            async_copy16(&smB[buf][col * KSTEP + kc],
                         Bt + (long)(colt + col) * K + kb + kc);
        }
    };

    v8f acc[4] = {v8f{}, v8f{}, v8f{}, v8f{}};
    const int nk = K / KSTEP;

    issue(0, 0);
    for (int i = 0; i < nk; ++i) {
        wait_async0();            // this wave's panel-i DMA has landed
        __syncthreads();          // all waves' panel i in LDS; all done panel i-1
        if (i + 1 < nk) issue((i + 1) * KSTEP, (i + 1) & 1);  // overlap DMA

        const unsigned short* pa = smA[i & 1];
        const unsigned short* pb = smB[i & 1];
#pragma unroll
        for (int kk = 0; kk < KSTEP; kk += 32) {
            FragBF af;
            const unsigned short* ar = pa + (16 * w + ln) * KSTEP + kk;
            af.q[0] = *(const v4u*)(ar +  0 + 8 * hf);
            af.q[1] = *(const v4u*)(ar + 16 + 8 * hf);
#pragma unroll
            for (int t = 0; t < 4; t++) {
                FragBF bfr;
                const unsigned short* br = pb + (16 * t + ln) * KSTEP + kk;
                bfr.q[0] = *(const v4u*)(br + 8 * hf);
                bfr.q[1] = *(const v4u*)(br + 16 + 8 * hf);
                acc[t] = __builtin_amdgcn_wmma_f32_16x16x32_bf16(
                    false, af.v, false, bfr.v, (short)0, acc[t], false, false);
            }
        }
    }

#pragma unroll
    for (int t = 0; t < 4; t++) {
        const int col = colt + 16 * t + ln;
        const float bv = bias ? bias[col] : 0.f;
#pragma unroll
        for (int r = 0; r < 8; r++) {
            float v = acc[t][r] + bv;
            if (relu) v = fmaxf(v, 0.f);
            const long idx = (long)(rowt + 16 * w + r + 8 * hf) * N + col;
            if (Cf) Cf[idx] = v;
            if (Cb) Cb[idx] = f2bf(v);
        }
    }
}

// ---------------------------------------------------------------------------
extern "C" void kernel_launch(void* const* d_in, const int* in_sizes, int n_in,
                              void* d_out, int out_size, void* d_ws, size_t ws_size,
                              hipStream_t stream) {
    (void)in_sizes; (void)n_in; (void)out_size; (void)ws_size;
    const float* src   = (const float*)d_in[0];
    const float* g1    = (const float*)d_in[1];
    const float* bln1  = (const float*)d_in[2];
    const float* g2    = (const float*)d_in[3];
    const float* bln2  = (const float*)d_in[4];
    const float* W1    = (const float*)d_in[5];
    const float* bias1 = (const float*)d_in[6];
    const float* W2    = (const float*)d_in[7];
    const float* bias2 = (const float*)d_in[8];
    float* out = (float*)d_out;

    char* ws = (char*)d_ws;
    size_t off = 0;
    auto carve = [&](size_t bytes) -> char* {
        char* p = ws + off;
        off += (bytes + 255) & ~(size_t)255;
        return p;
    };
    const long BS = (long)B_ * S_;
    float*          attn = (float*)carve(BS * DM * 4);
    float*          x    = (float*)carve(BS * DM * 4);
    unsigned short* xb   = (unsigned short*)carve(BS * DM * 2);
    unsigned short* qb   = (unsigned short*)carve((long)B_ * H_ * S_ * DH * 2);
    unsigned short* vt   = (unsigned short*)carve((long)B_ * H_ * S_ * DH * 2);
    unsigned short* w1t  = (unsigned short*)carve((long)DM * DFF * 2);
    unsigned short* w2t  = (unsigned short*)carve((long)DFF * DM * 2);
    unsigned short* hb   = (unsigned short*)carve(BS * DFF * 2);
    float*          ff   = (float*)carve(BS * DM * 4);

    prep_qkv<<<(B_ * S_ * DM) / 256, 256, 0, stream>>>(src, qb, vt);
    transpose_bf16<<<(DM * DFF) / 256, 256, 0, stream>>>(W1, w1t, DM, DFF);
    transpose_bf16<<<(DFF * DM) / 256, 256, 0, stream>>>(W2, w2t, DFF, DM);

    attn_kernel<<<B_ * H_ * NC, 512, 0, stream>>>(qb, vt, attn);

    ln_kernel<<<(int)BS, 256, 0, stream>>>(src, attn, g1, bln1, x, xb);

    gemm_bf16<<<dim3(DFF / 64, (int)(BS / 128)), 256, 0, stream>>>(
        xb, w1t, bias1, nullptr, hb, (int)BS, DFF, DM, 1);
    gemm_bf16<<<dim3(DM / 64, (int)(BS / 128)), 256, 0, stream>>>(
        hb, w2t, bias2, ff, nullptr, (int)BS, DM, DFF, 0);

    ln_kernel<<<(int)BS, 256, 0, stream>>>(x, ff, g2, bln2, out, nullptr);
}
